// RNN_76819785056270
// MI455X (gfx1250) — matrix operand order repaired and checked by reading, not compiled, not run on hardware
//
#include <hip/hip_runtime.h>
#include <hip/hip_bf16.h>

// Problem constants (from reference): S,B,V,E,H,L = 64,64,10000,512,512,2
#define S_  64
#define B_  64
#define V_  10000
#define E_  512
#define H_  512
#define K0_ (H_ + E_)   // 1024, layer-0 concat [h0; emb]
#define K1_ (2 * H_)    // 1024, layer-1 concat [h1; out]
#define NT_ (V_ / 16)   // 625 logits N-tiles (exact)

typedef __bf16 bf16;
typedef __attribute__((ext_vector_type(16))) bf16  v16bf;
typedef __attribute__((ext_vector_type(8)))  bf16  v8bf;
typedef __attribute__((ext_vector_type(8)))  float v8f;

// ---------------------------------------------------------------------------
// WMMA helpers — layouts used here (wave32); A/B K order is contiguous, not ISA table order
// A (16x32 bf16): lane m=lane&15, kh=lane>>4 holds K = kh*16..+15
//                 -> one contiguous 16-elem chunk of row m.
// B (32x16 bf16): lane n=lane&15, kh=lane>>4 holds K = kh*16..+15 of column n.
//                 We compute X @ W^T, so B[k][n] = W[n][k]: one contiguous
//                 16-elem chunk of row n of row-major W.
// C/D (16x16 f32): vgpr r, lane -> row r + 8*(lane>>4), col lane&15.
// ---------------------------------------------------------------------------
static __device__ __forceinline__ v8f wmma_bf16(v16bf a, v16bf b, v8f c) {
    return __builtin_amdgcn_wmma_f32_16x16x32_bf16(
        false, a, false, b, (short)0, c, false, false);
}

static __device__ __forceinline__ v16bf load_a_frag(const bf16* X, int ld,
                                                    int m0, int k, int lane) {
    const int m  = m0 + (lane & 15);
    const int kh = lane >> 4;
    const bf16* p = X + m * ld + k + kh * 16;
    return *(const v16bf*)(p);         // K = k+kh*16 .. +15
}

static __device__ __forceinline__ v16bf load_b_frag(const bf16* W, int ld,
                                                    int n0, int k, int lane) {
    const int n  = n0 + (lane & 15);
    const int kh = lane >> 4;
    return *(const v16bf*)(W + n * ld + k + kh * 16);   // 32B aligned
}

// ---------------------------------------------------------------------------
// Grid-wide sense barrier (8 co-resident blocks), s_sleep while spinning.
// ---------------------------------------------------------------------------
static __device__ __forceinline__ void grid_barrier(int* cnt, int* gen, int nblk) {
    __syncthreads();
    if (threadIdx.x == 0) {
        __threadfence();
        int g = *(volatile int*)gen;
        int a = atomicAdd(cnt, 1);
        if (a == nblk - 1) {
            *(volatile int*)cnt = 0;
            __threadfence();
            atomicAdd(gen, 1);
        } else {
            while (*(volatile int*)gen == g) { __builtin_amdgcn_s_sleep(2); }
        }
        __threadfence();
    }
    __syncthreads();
}

// ---------------------------------------------------------------------------
// Kernel 1: fp32 -> bf16 weight conversion + initial activation buffers.
// ---------------------------------------------------------------------------
__global__ void __launch_bounds__(256) prep_kernel(
    const float* __restrict__ W0, const float* __restrict__ Wr,
    const float* __restrict__ Wy, const float* __restrict__ hidden,
    const float* __restrict__ emb_table, const int* __restrict__ inputs,
    bf16* __restrict__ W0b, bf16* __restrict__ Wrb, bf16* __restrict__ Wyb,
    bf16* __restrict__ X0, bf16* __restrict__ X1)
{
    const int idx = blockIdx.x * blockDim.x + threadIdx.x;
    const int stride = gridDim.x * blockDim.x;

    for (int i = idx; i < V_ * H_; i += stride) Wyb[i] = (bf16)Wy[i];
    for (int i = idx; i < H_ * K0_; i += stride) W0b[i] = (bf16)W0[i];
    for (int i = idx; i < H_ * K1_; i += stride) Wrb[i] = (bf16)Wr[i];

    for (int i = idx; i < B_ * H_; i += stride) {
        int b = i >> 9, h = i & (H_ - 1);
        X0[b * K0_ + h] = (bf16)hidden[0 * B_ * H_ + i];   // h0(t=0)
        X1[b * K1_ + h] = (bf16)hidden[1 * B_ * H_ + i];   // h1(t=0)
    }
    for (int i = idx; i < B_ * E_; i += stride) {
        int b = i >> 9, e = i & (E_ - 1);
        int tok = inputs[b];                               // inputs[t=0][b]
        X0[b * K0_ + H_ + e] = (bf16)emb_table[tok * E_ + e];
    }
}

// ---------------------------------------------------------------------------
// Kernel 2: sequential recurrence. 8 blocks x 256 threads = 64 wave32s.
// Wave w: N-tile nt = w&31, M-tiles {2*(w>>5), 2*(w>>5)+1}: one B fragment
// per k step feeds two WMMAs (A varies). Weights re-read from L2 each step;
// an asm pointer-barrier inside the t-loop stops LICM from hoisting the
// (t-invariant) weight fragments into 2KB/lane scratch spills.
// ---------------------------------------------------------------------------
__global__ void __launch_bounds__(256, 1) rnn_recurrence_kernel(
    const int* __restrict__ inputs, const float* __restrict__ emb_table,
    const float* __restrict__ b0, const float* __restrict__ br,
    const bf16* __restrict__ W0b, const bf16* __restrict__ Wrb,
    bf16* __restrict__ X0, bf16* __restrict__ X1, bf16* __restrict__ Hseq,
    float* __restrict__ finalh, int* __restrict__ sync)
{
    const int lane = threadIdx.x & 31;
    const int tid  = blockIdx.x * blockDim.x + threadIdx.x;   // 0..2047
    const int wave = tid >> 5;                                // 0..63
    const int nt   = wave & 31;                               // N tile
    const int mt0  = (wave >> 5) * 2;                         // M tiles (2)
    const int mt1  = mt0 + 1;
    const int kh   = lane >> 4;
    const int nn   = lane & 15;
    const int col  = nt * 16 + nn;
    int* cnt = sync;
    int* gen = sync + 1;

    const float bias0 = b0[col];
    const float bias1 = br[col];

    for (int t = 0; t < S_; ++t) {
        // Opaque zero offset: compiler cannot prove weight loads t-invariant,
        // so it cannot hoist 2KB/lane of fragments out of this loop.
        unsigned zoff = 0;
        asm volatile("" : "+s"(zoff));
        const bf16* w0p = W0b + zoff;
        const bf16* wrp = Wrb + zoff;

        const bf16* X0c = X0 + (t & 1) * (B_ * K0_);
        bf16*       X0n = X0 + ((t + 1) & 1) * (B_ * K0_);
        bf16*       X1c = X1 + (t & 1) * (B_ * K1_);
        bf16*       X1n = X1 + ((t + 1) & 1) * (B_ * K1_);

        // ---- Phase 1: layer 0  out0 = tanh([h0|emb] @ W0^T + b0) ----
        v8f c0 = {}, c1 = {};
        for (int k = 0; k < K0_; k += 32) {
            v16bf b  = load_b_frag(w0p, K0_, nt * 16, k, lane);
            v16bf a0 = load_a_frag(X0c, K0_, mt0 * 16, k, lane);
            v16bf a1 = load_a_frag(X0c, K0_, mt1 * 16, k, lane);
            c0 = wmma_bf16(a0, b, c0);
            c1 = wmma_bf16(a1, b, c1);
        }
        #pragma unroll
        for (int r = 0; r < 8; ++r) {
            int row0 = mt0 * 16 + r + 8 * kh;         // batch indices
            int row1 = mt1 * 16 + r + 8 * kh;
            float v0 = tanhf(c0[r] + bias0);
            float v1 = tanhf(c1[r] + bias0);
            X1c[row0 * K1_ + H_ + col] = (bf16)v0;    // feeds layer-1 concat
            X1c[row1 * K1_ + H_ + col] = (bf16)v1;
            X0n[row0 * K0_ + col] = (bf16)v0;         // next-step h0
            X0n[row1 * K0_ + col] = (bf16)v1;
            if (t == S_ - 1) {                        // final_hidden[0]
                finalh[row0 * H_ + col] = v0;
                finalh[row1 * H_ + col] = v1;
            }
        }
        // embedding gather for step t+1 into X0(next)[:,H:]
        if (t + 1 < S_) {
            for (int i = tid; i < B_ * E_; i += 2048) {
                int b = i >> 9, e = i & (E_ - 1);
                int tok = inputs[(t + 1) * B_ + b];
                X0n[b * K0_ + H_ + e] = (bf16)emb_table[tok * E_ + e];
            }
        }
        grid_barrier(cnt, gen, gridDim.x);

        // ---- Phase 2: layer 1  out1 = tanh([h1|out0] @ Wr^T + br) ----
        v8f d0 = {}, d1 = {};
        for (int k = 0; k < K1_; k += 32) {
            v16bf b  = load_b_frag(wrp, K1_, nt * 16, k, lane);
            v16bf a0 = load_a_frag(X1c, K1_, mt0 * 16, k, lane);
            v16bf a1 = load_a_frag(X1c, K1_, mt1 * 16, k, lane);
            d0 = wmma_bf16(a0, b, d0);
            d1 = wmma_bf16(a1, b, d1);
        }
        #pragma unroll
        for (int r = 0; r < 8; ++r) {
            int row0 = mt0 * 16 + r + 8 * kh;
            int row1 = mt1 * 16 + r + 8 * kh;
            float v0 = tanhf(d0[r] + bias1);
            float v1 = tanhf(d1[r] + bias1);
            X1n[row0 * K1_ + col] = (bf16)v0;         // next-step h1
            X1n[row1 * K1_ + col] = (bf16)v1;
            Hseq[(t * B_ + row0) * H_ + col] = (bf16)v0;
            Hseq[(t * B_ + row1) * H_ + col] = (bf16)v1;
            if (t == S_ - 1) {                        // final_hidden[1]
                finalh[B_ * H_ + row0 * H_ + col] = v0;
                finalh[B_ * H_ + row1 * H_ + col] = v1;
            }
        }
        grid_barrier(cnt, gen, gridDim.x);
    }
}

// ---------------------------------------------------------------------------
// Kernel 3: logits = Hseq(4096x512) @ Wy^T(512x10000) + by, fp32 out.
// Block: 64 M-rows staged into LDS (64KB) with gfx1250 async DMA loads
// (GLOBAL_LOAD_ASYNC_TO_LDS_B128, ASYNCcnt-tracked) x 16 N-tiles.
// Wave: 4 M-tiles x 2 N-tiles -> 8 accumulators; B fragments reused 4x from
// registers, A fragments come from LDS (ds_load). Wy bf16 (10MB) is
// L2-resident; prefetch next K-slice.
// ---------------------------------------------------------------------------
__global__ void __launch_bounds__(256, 2) logits_gemm_kernel(
    const bf16* __restrict__ Hseq, const bf16* __restrict__ Wyb,
    const float* __restrict__ by, float* __restrict__ out)
{
    extern __shared__ char smem_raw[];
    bf16* As = (bf16*)smem_raw;                       // [64][512] bf16 = 64KB

    const int lane = threadIdx.x & 31;
    const int wv   = threadIdx.x >> 5;                // 0..7
    const int m0   = blockIdx.y * 64;

    // Async-DMA stage of the A block (all 8 waves share it). The dynamic
    // extern __shared__ region is the only LDS allocation, so its
    // wave-relative LDS offset is 0: LDS destinations are plain byte offsets.
    // 16 iters/thread x 256 threads x 16B = 64KB per block.
    {
        const char* src = (const char*)(Hseq + (size_t)m0 * H_);
        #pragma unroll 4
        for (int i = threadIdx.x; i < (64 * H_ * 2) / 16; i += 256) {
            unsigned    lds_off = (unsigned)i * 16u;
            const char* gaddr   = src + (size_t)i * 16;
            asm volatile("global_load_async_to_lds_b128 %0, %1, off"
                         :: "v"(lds_off), "v"(gaddr) : "memory");
        }
        asm volatile("s_wait_asynccnt 0x0" ::: "memory");
    }
    __syncthreads();

    const int ntA = (blockIdx.x * 8 + wv) * 2;        // two adjacent N-tiles
    const int ntB = ntA + 1;
    if (ntA >= NT_) return;                           // after the barrier
    const bool okB = (ntB < NT_);
    const int n0A = ntA * 16;
    const int n0B = (okB ? ntB : ntA) * 16;           // clamp loads in-range

    v8f acc[8] = {{}, {}, {}, {}, {}, {}, {}, {}};
    for (int k = 0; k < H_; k += 32) {
        v16bf bA = load_b_frag(Wyb, H_, n0A, k, lane);
        v16bf bB = load_b_frag(Wyb, H_, n0B, k, lane);
        if (k + 64 < H_) {  // pull next Wy K-slice toward the WGP
            __builtin_prefetch((const void*)(Wyb + (n0A + (lane & 15)) * H_ + k + 64), 0, 0);
        }
        #pragma unroll
        for (int i = 0; i < 4; ++i) {
            v16bf a = load_a_frag(As, H_, i * 16, k, lane);   // ds_load path
            acc[2 * i + 0] = wmma_bf16(a, bA, acc[2 * i + 0]);
            acc[2 * i + 1] = wmma_bf16(a, bB, acc[2 * i + 1]);
        }
    }
    const int kh = lane >> 4, nn = lane & 15;
    const float biasA = by[n0A + nn];
    const float biasB = by[n0B + nn];
    #pragma unroll
    for (int i = 0; i < 4; ++i) {
        #pragma unroll
        for (int r = 0; r < 8; ++r) {
            int row = m0 + i * 16 + r + 8 * kh;       // row = t*B + b
            out[(size_t)row * V_ + n0A + nn] = acc[2 * i + 0][r] + biasA;
            if (okB)
                out[(size_t)row * V_ + n0B + nn] = acc[2 * i + 1][r] + biasB;
        }
    }
}

// ---------------------------------------------------------------------------
// Host-side orchestration (graph-capture safe: async ops on `stream` only).
// ---------------------------------------------------------------------------
extern "C" void kernel_launch(void* const* d_in, const int* in_sizes, int n_in,
                              void* d_out, int out_size, void* d_ws, size_t ws_size,
                              hipStream_t stream) {
    const int*   inputs    = (const int*)  d_in[0];   // [S,B] int32
    const float* hidden    = (const float*)d_in[1];   // [L,B,H]
    const float* emb_table = (const float*)d_in[2];   // [V,E]
    const float* W0        = (const float*)d_in[3];   // [H,H+E]
    const float* b0        = (const float*)d_in[4];   // [H]
    const float* Wr        = (const float*)d_in[5];   // [1,H,2H]
    const float* br        = (const float*)d_in[6];   // [1,H]
    const float* Wy        = (const float*)d_in[7];   // [V,H]
    const float* by        = (const float*)d_in[8];   // [V]

    float* logits = (float*)d_out;                            // [S,B,V]
    float* finalh = (float*)d_out + (size_t)S_ * B_ * V_;     // [L,B,H]

    // Workspace layout (256B-aligned chunks)
    char* ws = (char*)d_ws;
    size_t off = 0;
    int*  sync = (int*)(ws + off);            off += 256;
    bf16* W0b  = (bf16*)(ws + off);           off += (size_t)H_ * K0_ * 2;
    bf16* Wrb  = (bf16*)(ws + off);           off += (size_t)H_ * K1_ * 2;
    bf16* Wyb  = (bf16*)(ws + off);           off += (size_t)V_ * H_ * 2;
    bf16* X0   = (bf16*)(ws + off);           off += (size_t)2 * B_ * K0_ * 2;
    bf16* X1   = (bf16*)(ws + off);           off += (size_t)2 * B_ * K1_ * 2;
    bf16* Hseq = (bf16*)(ws + off);           off += (size_t)S_ * B_ * H_ * 2;
    (void)ws_size; (void)in_sizes; (void)n_in; (void)out_size;

    // Barrier state must be zero each call (ws is poisoned before timing).
    hipMemsetAsync(sync, 0, 256, stream);

    prep_kernel<<<2048, 256, 0, stream>>>(W0, Wr, Wy, hidden, emb_table, inputs,
                                          W0b, Wrb, Wyb, X0, X1);

    rnn_recurrence_kernel<<<8, 256, 0, stream>>>(inputs, emb_table, b0, br,
                                                 W0b, Wrb, X0, X1, Hseq,
                                                 finalh, sync);

    dim3 lgrid((NT_ + 15) / 16, (S_ * B_) / 64);     // (40, 64)
    logits_gemm_kernel<<<lgrid, 256, 64 * H_ * 2, stream>>>(Hseq, Wyb, by, logits);
}
